// _SelectiveSSMBlock_58815282152002
// MI455X (gfx1250) — matrix operand order repaired and checked by reading, not compiled, not run on hardware
//
#include <hip/hip_runtime.h>
#include <hip/hip_bf16.h>

// ---------------------------------------------------------------------------
// Selective SSM (Mamba-style) block for gfx1250 / MI455X.
// Large GEMMs: 128x64 block tile, async-to-LDS double-buffered K-panels,
//              32x32 per wave => 4 independent v_wmma_f32_16x16x32_bf16 / step.
// ---------------------------------------------------------------------------

typedef __bf16 bf16;
typedef __attribute__((ext_vector_type(16))) __bf16 v16bf;
typedef __attribute__((ext_vector_type(8)))  __bf16 v8bf;
typedef __attribute__((ext_vector_type(8)))  float  v8f;

#define BATCH   2
#define SEQ     2048
#define DM      768
#define DSTATE  16
#define NIN     (2*DM)          // 1536
#define MROWS   (BATCH*SEQ)     // 4096

// ---------------------------------------------------------------------------
// CDNA5 async global->LDS copy (16B per lane), ASYNCcnt-tracked.
// ---------------------------------------------------------------------------
__device__ __forceinline__ void async_copy16(unsigned lds_off, const bf16* g) {
    asm volatile("global_load_async_to_lds_b128 %0, %1, off"
                 :: "v"(lds_off), "v"(g) : "memory");
}
__device__ __forceinline__ void wait_async0() {
    asm volatile("s_wait_asynccnt 0x0" ::: "memory");
}
__device__ __forceinline__ void wait_async3() {
    asm volatile("s_wait_asynccnt 0x3" ::: "memory");
}

// ---------------------------------------------------------------------------
// f32 -> bf16 convert
// ---------------------------------------------------------------------------
__global__ void k_cvt_bf16(const float* __restrict__ in, bf16* __restrict__ out, int n) {
    int i = blockIdx.x * blockDim.x + threadIdx.x;
    if (i < n) out[i] = (bf16)in[i];
}

// ---------------------------------------------------------------------------
// Build the 16-bit A-fragment (ISA 16x32 layout) from a row-major K-panel:
// elems 0-7 at col kh*8, elems 8-15 at col kh*8+16 (32 bf16 cols per row).
// ---------------------------------------------------------------------------
__device__ __forceinline__ v16bf frag_from_panel(const bf16* p /*row base + kh*8*/) {
    const v8bf* q = reinterpret_cast<const v8bf*>(p);
    v8bf lo = q[0];
    v8bf hi = q[2];                 // +16 elements
    v16bf a;
#pragma unroll
    for (int i = 0; i < 8; ++i) { a[i] = lo[i]; a[i + 8] = hi[i]; }
    return a;
}

// ---------------------------------------------------------------------------
// Staged WMMA GEMM: C[M,N] = A[M,K] * W[N,K]^T   (bf16 in, f32 out)
// Block = 128(M) x 64(N); 8 waves as 4x2; wave = 32x32 = 2x2 WMMA tiles.
// K-panels (A 128x32, B 64x32) double-buffered in LDS via async loads.
// Requires: M % 128 == 0, N % 64 == 0, K % 32 == 0, K >= 32.
// ---------------------------------------------------------------------------
__global__ void __launch_bounds__(256)
k_gemm_bf16_staged(const bf16* __restrict__ A, const bf16* __restrict__ W,
                   float* __restrict__ C, int Mr, int Nc, int Kd) {
    __shared__ __align__(64) bf16 sA[2][128 * 32];   // 16 KB
    __shared__ __align__(64) bf16 sB[2][64 * 32];    //  8 KB

    const int tid  = threadIdx.x;
    const int lane = tid & 31;
    const int wave = tid >> 5;
    const int l15  = lane & 15;
    const int kh   = lane >> 4;          // K-half select
    const int wm   = wave >> 1;          // 0..3 (M)
    const int wn   = wave & 1;           // 0..1 (N)

    const int tilesN = Nc >> 6;
    const int bm = blockIdx.x / tilesN;
    const int bn = blockIdx.x - bm * tilesN;
    const int m0 = bm << 7;
    const int n0 = bn << 6;

    // copy-role: thread tid moves three 16B chunks per panel
    const int crow = tid >> 2;                 // 0..63
    const int ccol = (tid & 3) << 3;           // bf16 col: 0,8,16,24
    const bf16* gA0 = A + (size_t)(m0 + crow)      * Kd + ccol;
    const bf16* gA1 = A + (size_t)(m0 + crow + 64) * Kd + ccol;
    const bf16* gB  = W + (size_t)(n0 + crow)      * Kd + ccol;
    const unsigned coff = (unsigned)(crow * 64 + ((tid & 3) << 4));   // bytes
    unsigned dA[2], dB[2];
#pragma unroll
    for (int b = 0; b < 2; ++b) {
        dA[b] = (unsigned)(uintptr_t)(&sA[b][0]) + coff;
        dB[b] = (unsigned)(uintptr_t)(&sB[b][0]) + coff;
    }

    v8f acc00 = {}, acc01 = {}, acc10 = {}, acc11 = {};

    auto issue = [&](int b, int kk) {
        async_copy16(dA[b],           gA0 + kk);
        async_copy16(dA[b] + 64 * 64, gA1 + kk);
        async_copy16(dB[b],           gB  + kk);
    };
    auto compute = [&](int b) {
        const bf16* pA = &sA[b][0];
        const bf16* pB = &sB[b][0];
        v16bf a0 = frag_from_panel(pA + (wm * 32 +      l15) * 32 + kh * 8);
        v16bf a1 = frag_from_panel(pA + (wm * 32 + 16 + l15) * 32 + kh * 8);
        v16bf b0 = frag_from_panel(pB + (wn * 32 +      l15) * 32 + kh * 8);
        v16bf b1 = frag_from_panel(pB + (wn * 32 + 16 + l15) * 32 + kh * 8);
        acc00 = __builtin_amdgcn_wmma_f32_16x16x32_bf16(false, a0, false, b0, (short)0, acc00, false, false);
        acc01 = __builtin_amdgcn_wmma_f32_16x16x32_bf16(false, a0, false, b1, (short)0, acc01, false, false);
        acc10 = __builtin_amdgcn_wmma_f32_16x16x32_bf16(false, a1, false, b0, (short)0, acc10, false, false);
        acc11 = __builtin_amdgcn_wmma_f32_16x16x32_bf16(false, a1, false, b1, (short)0, acc11, false, false);
    };

    // prologue: issue panel 0
    issue(0, 0);

    const int nK = Kd >> 5;
    int buf = 0;
    // steady state: branch-free (always a next panel to issue)
    for (int ki = 0; ki < nK - 1; ++ki) {
        issue(buf ^ 1, (ki + 1) << 5);
        wait_async3();              // current panel's 3 ops retired (in-order)
        __syncthreads();            // all waves' panels visible
        compute(buf);
        __syncthreads();            // safe to overwrite this buffer next round
        buf ^= 1;
    }
    // epilogue: last panel
    wait_async0();
    __syncthreads();
    compute(buf);

    // C/D layout: VGPR r -> row (kh*8 + r), col l15 within each 16x16 tile
    const int rbase = m0 + wm * 32 + kh * 8;
    const int cbase = n0 + wn * 32 + l15;
#pragma unroll
    for (int r = 0; r < 8; ++r) {
        float* c0 = C + (size_t)(rbase + r) * Nc + cbase;
        float* c1 = C + (size_t)(rbase + 16 + r) * Nc + cbase;
        c0[0]  = acc00[r];
        c0[16] = acc01[r];
        c1[0]  = acc10[r];
        c1[16] = acc11[r];
    }
}

// ---------------------------------------------------------------------------
// Simple WMMA GEMM (16x16 per wave) for the small N=32 projection.
// ---------------------------------------------------------------------------
__global__ void k_gemm_bf16(const bf16* __restrict__ A, const bf16* __restrict__ W,
                            float* __restrict__ C, int Mr, int Nc, int Kd) {
    const int lane = threadIdx.x & 31;
    const int wave = threadIdx.x >> 5;
    const int tilesN = Nc >> 4;
    const int tile = blockIdx.x * 8 + wave;
    if (tile >= (Mr >> 4) * tilesN) return;   // wave-uniform
    const int tm = (tile / tilesN) << 4;
    const int tn = (tile % tilesN) << 4;
    const int l15 = lane & 15;
    const int kh  = lane >> 4;

    const bf16* __restrict__ arow = A + (size_t)(tm + l15) * Kd + kh * 8;
    const bf16* __restrict__ brow = W + (size_t)(tn + l15) * Kd + kh * 8;

    v8f acc = {};
    for (int kk = 0; kk < Kd; kk += 32) {
        v16bf a = frag_from_panel(arow + kk);   // same gather, global memory
        v16bf b = frag_from_panel(brow + kk);
        if (kk + 32 < Kd) __builtin_prefetch(brow + kk + 32, 0, 1);
        acc = __builtin_amdgcn_wmma_f32_16x16x32_bf16(false, a, false, b, (short)0, acc, false, false);
    }
    float* crow = C + (size_t)(tm + kh * 8) * Nc + tn + l15;
#pragma unroll
    for (int r = 0; r < 8; ++r) crow[(size_t)r * Nc] = acc[r];
}

// ---------------------------------------------------------------------------
// Depthwise conv (taps t-3..t) + bias + SiLU -> u (f32) and u (bf16)
// ---------------------------------------------------------------------------
__global__ void k_conv_silu(const float* __restrict__ xz,
                            const float* __restrict__ cw,
                            const float* __restrict__ cb,
                            float* __restrict__ u, bf16* __restrict__ ub) {
    int idx = blockIdx.x * blockDim.x + threadIdx.x;
    if (idx >= MROWS * DM) return;
    int r = idx / DM;           // b*SEQ + t
    int d = idx - r * DM;
    int t = r & (SEQ - 1);
    float acc = cb[d];
#pragma unroll
    for (int k = 0; k < 4; ++k) {
        int tt = t - 3 + k;
        if (tt >= 0)
            acc += xz[(size_t)(r - 3 + k) * NIN + d] * cw[d * 4 + k];
    }
    float s = acc / (1.f + __expf(-acc));
    u[idx]  = s;
    ub[idx] = (bf16)s;
}

// ---------------------------------------------------------------------------
// dt = softplus(v + dt_b)  in place
// ---------------------------------------------------------------------------
__global__ void k_bias_softplus(float* __restrict__ v, const float* __restrict__ b, int n) {
    int i = blockIdx.x * blockDim.x + threadIdx.x;
    if (i >= n) return;
    float x = v[i] + b[i % DM];
    v[i] = (x > 20.f) ? x : log1pf(__expf(x));
}

// ---------------------------------------------------------------------------
// Selective scan. Thread per (b,d,n); h in a register; y reduced over the
// 16-lane state group with shfl_xor. Software-pipelined: loads for t+1 are
// issued before the serial exp/fma chain of step t (branch-free steady state).
// ---------------------------------------------------------------------------
__global__ void k_scan(const float* __restrict__ dt, const float* __restrict__ u,
                       const float* __restrict__ bc, const float* __restrict__ xz,
                       const float* __restrict__ A_log, const float* __restrict__ Dp,
                       bf16* __restrict__ y1b) {
    int tid = blockIdx.x * blockDim.x + threadIdx.x;   // 0..24575
    int n  = tid & 15;
    int ch = tid >> 4;
    int b  = ch / DM;
    int d  = ch - b * DM;
    float Ac = -__expf(A_log[d * DSTATE + n]);
    float Dd = Dp[d];
    float h  = 0.f;
    size_t r = (size_t)b * SEQ;

    float dtv = dt[r * DM + d];
    float uv  = u [r * DM + d];
    float bv  = bc[r * 32 + n];
    float cv  = bc[r * 32 + 16 + n];

    auto step = [&](float dtc, float uc, float bcur, float ccur) {
        h = __expf(dtc * Ac) * h + dtc * bcur * uc;
        float p = h * ccur;
        p += __shfl_xor(p, 1);
        p += __shfl_xor(p, 2);
        p += __shfl_xor(p, 4);
        p += __shfl_xor(p, 8);
        if (n == 0) {
            float zv = xz[r * NIN + DM + d];
            float yv = (p + Dd * uc) * (zv / (1.f + __expf(-zv)));
            y1b[r * DM + d] = (bf16)yv;
        }
    };

    for (int t = 0; t < SEQ - 1; ++t) {
        // unconditional prefetch of step t+1 (off the recurrence chain)
        size_t r2  = r + 1;
        float dtn  = dt[r2 * DM + d];
        float un   = u [r2 * DM + d];
        float bn2  = bc[r2 * 32 + n];
        float cn2  = bc[r2 * 32 + 16 + n];
        step(dtv, uv, bv, cv);
        dtv = dtn; uv = un; bv = bn2; cv = cn2;
        ++r;
    }
    step(dtv, uv, bv, cv);          // final step, nothing to prefetch
}

// ---------------------------------------------------------------------------
// Residual add + LayerNorm. One 256-thread block per (b,t) row.
// ---------------------------------------------------------------------------
__global__ void k_ln(const float* __restrict__ yo, const float* __restrict__ x,
                     const float* __restrict__ g, const float* __restrict__ bb,
                     float* __restrict__ out) {
    __shared__ float s1[256];
    __shared__ float s2[256];
    const int row = blockIdx.x;
    const int tx  = threadIdx.x;
    float v[3], sum = 0.f, sq = 0.f;
#pragma unroll
    for (int i = 0; i < 3; ++i) {
        int c = tx + i * 256;
        float val = yo[(size_t)row * DM + c] + x[(size_t)row * DM + c];
        v[i] = val; sum += val; sq += val * val;
    }
    s1[tx] = sum; s2[tx] = sq; __syncthreads();
    for (int st = 128; st > 0; st >>= 1) {
        if (tx < st) { s1[tx] += s1[tx + st]; s2[tx] += s2[tx + st]; }
        __syncthreads();
    }
    float mu   = s1[0] * (1.f / DM);
    float var  = s2[0] * (1.f / DM) - mu * mu;
    float rstd = rsqrtf(var + 1e-5f);
#pragma unroll
    for (int i = 0; i < 3; ++i) {
        int c = tx + i * 256;
        out[(size_t)row * DM + c] = (v[i] - mu) * rstd * g[c] + bb[c];
    }
}

// ---------------------------------------------------------------------------
extern "C" void kernel_launch(void* const* d_in, const int* in_sizes, int n_in,
                              void* d_out, int out_size, void* d_ws, size_t ws_size,
                              hipStream_t stream) {
    const float* x        = (const float*)d_in[0];
    const float* in_proj  = (const float*)d_in[1];
    const float* conv_w   = (const float*)d_in[2];
    const float* conv_b   = (const float*)d_in[3];
    const float* dt_w     = (const float*)d_in[4];
    const float* dt_b     = (const float*)d_in[5];
    const float* B_w      = (const float*)d_in[6];
    const float* C_w      = (const float*)d_in[7];
    const float* A_log    = (const float*)d_in[8];
    const float* D_param  = (const float*)d_in[9];
    const float* out_w    = (const float*)d_in[10];
    const float* ln_g     = (const float*)d_in[11];
    const float* ln_b     = (const float*)d_in[12];
    float* out = (float*)d_out;

    char*  ws = (char*)d_ws;
    size_t o  = 0;
    auto take = [&](size_t bytes) { size_t r = o; o = (o + bytes + 255) & ~(size_t)255; return r; };
    bf16*  xb    = (bf16*) (ws + take((size_t)MROWS * DM * 2));
    bf16*  wib   = (bf16*) (ws + take((size_t)NIN * DM * 2));
    bf16*  wdtb  = (bf16*) (ws + take((size_t)DM * DM * 2));
    bf16*  wob   = (bf16*) (ws + take((size_t)DM * DM * 2));
    bf16*  wbcb  = (bf16*) (ws + take((size_t)32 * DM * 2));
    float* xz    = (float*)(ws + take((size_t)MROWS * NIN * 4));
    float* u     = (float*)(ws + take((size_t)MROWS * DM * 4));
    bf16*  ub    = (bf16*) (ws + take((size_t)MROWS * DM * 2));
    float* dtbuf = (float*)(ws + take((size_t)MROWS * DM * 4));
    float* bc    = (float*)(ws + take((size_t)MROWS * 32 * 4));
    bf16*  y1b   = (bf16*) (ws + take((size_t)MROWS * DM * 2));
    float* yo    = (float*)(ws + take((size_t)MROWS * DM * 4));
    (void)ws_size; (void)in_sizes; (void)n_in; (void)out_size;

    const int BT = 256;
    auto g1 = [&](int n) { return (n + BT - 1) / BT; };

    // converts
    k_cvt_bf16<<<g1(MROWS * DM), BT, 0, stream>>>(x, xb, MROWS * DM);
    k_cvt_bf16<<<g1(NIN * DM),  BT, 0, stream>>>(in_proj, wib, NIN * DM);
    k_cvt_bf16<<<g1(DM * DM),   BT, 0, stream>>>(dt_w, wdtb, DM * DM);
    k_cvt_bf16<<<g1(DM * DM),   BT, 0, stream>>>(out_w, wob, DM * DM);
    k_cvt_bf16<<<g1(DSTATE * DM), BT, 0, stream>>>(B_w, wbcb, DSTATE * DM);
    k_cvt_bf16<<<g1(DSTATE * DM), BT, 0, stream>>>(C_w, wbcb + DSTATE * DM, DSTATE * DM);

    // xz = x @ in_proj^T
    k_gemm_bf16_staged<<<(MROWS/128)*(NIN/64), BT, 0, stream>>>(xb, wib, xz, MROWS, NIN, DM);

    // conv + SiLU
    k_conv_silu<<<g1(MROWS * DM), BT, 0, stream>>>(xz, conv_w, conv_b, u, ub);

    // dt = softplus(u @ dt_w^T + dt_b)
    k_gemm_bf16_staged<<<(MROWS/128)*(DM/64), BT, 0, stream>>>(ub, wdtb, dtbuf, MROWS, DM, DM);
    k_bias_softplus<<<g1(MROWS * DM), BT, 0, stream>>>(dtbuf, dt_b, MROWS * DM);

    // [Bv|Cv] = u @ [B_w;C_w]^T  (N = 32, simple kernel)
    k_gemm_bf16<<<(MROWS/16)*(32/16)/8, BT, 0, stream>>>(ub, wbcb, bc, MROWS, 32, DM);

    // selective scan (+D*u, *silu(z)) -> bf16
    k_scan<<<(BATCH * DM * DSTATE) / BT, BT, 0, stream>>>(dtbuf, u, bc, xz, A_log, D_param, y1b);

    // yo = y1 @ out_w^T
    k_gemm_bf16_staged<<<(MROWS/128)*(DM/64), BT, 0, stream>>>(y1b, wob, yo, MROWS, DM, DM);

    // residual + LayerNorm
    k_ln<<<MROWS, BT, 0, stream>>>(yo, x, ln_g, ln_b, out);
}
